// SelfAttentiveLBLBiLM_3075196584309
// MI455X (gfx1250) — matrix operand order repaired and checked by reading, not compiled, not run on hardware
//
#include <hip/hip_runtime.h>

// ---------------- problem constants (from reference) ----------------
#define B_   16
#define L_   512
#define H_   512
#define NH_  8
#define NL_  2
#define NHW_ 2
#define W_   16
#define DK_  64
#define S_   (L_ + 2 * W_)   // 544
#define H2_  (2 * H_)        // 1024
#define H3_  (3 * H_)        // 1536

// ---------------- types for WMMA ----------------
typedef __bf16 bf16_t;
typedef bf16_t bf16x16 __attribute__((ext_vector_type(16)));
typedef float  f32x8   __attribute__((ext_vector_type(8)));

__device__ __forceinline__ unsigned f2bf(float f) {
  unsigned u = __float_as_uint(f);
  u += 0x7FFFu + ((u >> 16) & 1u);   // round-to-nearest-even
  return u >> 16;
}

// =====================================================================
// f32 -> bf16 conversion (8 elements / thread); n must be multiple of 2048
// =====================================================================
__global__ __launch_bounds__(256) void cvt_bf16(
    const float* __restrict__ in, unsigned short* __restrict__ out)
{
  size_t i = ((size_t)blockIdx.x * 256 + threadIdx.x) * 8;
  float4 v0 = *(const float4*)(in + i);
  float4 v1 = *(const float4*)(in + i + 4);
  uint4 p;
  p.x = f2bf(v0.x) | (f2bf(v0.y) << 16);
  p.y = f2bf(v0.z) | (f2bf(v0.w) << 16);
  p.z = f2bf(v1.x) | (f2bf(v1.y) << 16);
  p.w = f2bf(v1.z) | (f2bf(v1.w) << 16);
  *(uint4*)(out + i) = p;
}

// =====================================================================
// GEMM: C[M,N](f32) = A[M,K](bf16) * W[N,K]^T(bf16) + bias[N]
// Block tile 128x128, BK=32, 256 thr = 8 waves (4 along M x 2 along N);
// each wave owns 32x64 -> 8 accumulators, 8 wmma per K-step.
// Double-buffered LDS; one barrier per K-step.
// =====================================================================
#define BK 32
#define LDSS 40   // halves per LDS row (BK + 8 pad, 80B; 16B-aligned frag reads)

__global__ __launch_bounds__(256) void gemm_bias_bf16(
    const bf16_t* __restrict__ A, const bf16_t* __restrict__ Wt,
    const float* __restrict__ bias, float* __restrict__ C,
    int M, int N, int K)
{
  __shared__ bf16_t sA[2][128 * LDSS];
  __shared__ bf16_t sB[2][128 * LDSS];

  const int tid  = threadIdx.x;
  const int lane = tid & 31;
  const int wave = tid >> 5;
  const int wm   = wave & 3;    // 4 waves along M (32 rows each)
  const int wn   = wave >> 2;   // 2 waves along N (64 cols each)
  const int tileM = blockIdx.x * 128;
  const int tileN = blockIdx.y * 128;

  // staging: thread handles rows row0 and row0+64, 8 halves at off0
  const int row0 = tid >> 2;          // 0..63
  const int off0 = (tid & 3) * 8;     // 0,8,16,24
  const bf16_t* ag = A  + (size_t)(tileM + row0) * K + off0;
  const bf16_t* wg = Wt + (size_t)(tileN + row0) * K + off0;
  const size_t rowK64 = (size_t)64 * K;
  const int s0 = row0 * LDSS + off0;
  const int s1 = (row0 + 64) * LDSS + off0;

  f32x8 acc[2][4] = {};

  // prologue: stage K-tile 0 into buffer 0
  {
    uint4 a0 = *(const uint4*)(ag);
    uint4 a1 = *(const uint4*)(ag + rowK64);
    uint4 b0 = *(const uint4*)(wg);
    uint4 b1 = *(const uint4*)(wg + rowK64);
    *(uint4*)(&sA[0][s0]) = a0;
    *(uint4*)(&sA[0][s1]) = a1;
    *(uint4*)(&sB[0][s0]) = b0;
    *(uint4*)(&sB[0][s1]) = b1;
  }
  __syncthreads();

  const int rsel = lane & 15;
  const int akb  = (lane >> 4) * 8;
  const int bkb  = (lane >> 4) * 8;

  int buf = 0;
  for (int kb = 0; kb < K; kb += BK) {
    const bool nxt = (kb + BK) < K;
    uint4 na0, na1, nb0, nb1;
    if (nxt) {                              // overlap global loads with wmma
      na0 = *(const uint4*)(ag + kb + BK);
      na1 = *(const uint4*)(ag + rowK64 + kb + BK);
      nb0 = *(const uint4*)(wg + kb + BK);
      nb1 = *(const uint4*)(wg + rowK64 + kb + BK);
    }

    // A fragments (16x32): lanes 0-15 hold K {0..7,16..23}; 16-31 {8..15,24..31}
    union F { bf16x16 v; uint4 q[2]; };
    F fa[2], fb[4];
#pragma unroll
    for (int i2 = 0; i2 < 2; ++i2) {
      const bf16_t* ap = &sA[buf][(wm * 32 + i2 * 16 + rsel) * LDSS + akb];
      fa[i2].q[0] = *(const uint4*)(ap);
      fa[i2].q[1] = *(const uint4*)(ap + 16);
    }
    // B fragments (32x16): lane = N col, K {0..7,16..23} (lanes 0-15) / {8..15,24..31} (16-31)
#pragma unroll
    for (int g = 0; g < 4; ++g) {
      const bf16_t* bp = &sB[buf][(wn * 64 + g * 16 + rsel) * LDSS + bkb];
      fb[g].q[0] = *(const uint4*)(bp);
      fb[g].q[1] = *(const uint4*)(bp + 16);
    }
#pragma unroll
    for (int i2 = 0; i2 < 2; ++i2)
#pragma unroll
      for (int g = 0; g < 4; ++g)
        acc[i2][g] = __builtin_amdgcn_wmma_f32_16x16x32_bf16(
            false, fa[i2].v, false, fb[g].v, (short)0, acc[i2][g], false, false);

    if (nxt) {
      *(uint4*)(&sA[buf ^ 1][s0]) = na0;
      *(uint4*)(&sA[buf ^ 1][s1]) = na1;
      *(uint4*)(&sB[buf ^ 1][s0]) = nb0;
      *(uint4*)(&sB[buf ^ 1][s1]) = nb1;
      __syncthreads();
      buf ^= 1;
    }
  }

  // D layout: VGPR r -> M = r (lanes 0-15) / 8+r (lanes 16-31); N = lane&15
#pragma unroll
  for (int i2 = 0; i2 < 2; ++i2) {
    const int crow = tileM + wm * 32 + i2 * 16 + ((lane >> 4) * 8);
#pragma unroll
    for (int g = 0; g < 4; ++g) {
      const int ccol = tileN + wn * 64 + g * 16 + (lane & 15);
      const float bv = bias[ccol];
#pragma unroll
      for (int r = 0; r < 8; ++r)
        C[(size_t)(crow + r) * N + ccol] = acc[i2][g][r] + bv;
    }
  }
}

// =====================================================================
// pad: xp_bf16(B,S,H) = [pad_f(W,H) | x(B,L,H) | pad_b(W,H)]
// =====================================================================
__global__ __launch_bounds__(256) void pad_concat(
    const float* __restrict__ x, const float* __restrict__ padf,
    const float* __restrict__ padb, unsigned short* __restrict__ xp)
{
  size_t idx = (size_t)blockIdx.x * 256 + threadIdx.x;
  int h = idx % H_;
  int s = (idx / H_) % S_;
  int b = idx / ((size_t)H_ * S_);
  float v;
  if (s < W_)            v = padf[s * H_ + h];
  else if (s < W_ + L_)  v = x[((size_t)b * L_ + (s - W_)) * H_ + h];
  else                   v = padb[(s - W_ - L_) * H_ + h];
  xp[idx] = (unsigned short)f2bf(v);
}

// =====================================================================
// banded attention: one wave per (b, head, query). qkv = (B*S, 1536) f32:
// q at col h*64+d, k at 512+..., v at 1024+... Band of 18 keys.
// Output written as bf16 (feeds the O-projection GEMM only).
// =====================================================================
__global__ __launch_bounds__(256) void band_attn(
    const float* __restrict__ qkv, unsigned short* __restrict__ o, int isBwd)
{
  int gwave = (int)((blockIdx.x * 256 + threadIdx.x) >> 5);
  int lane  = threadIdx.x & 31;
  int i = gwave % S_;
  int h = (gwave / S_) % NH_;
  int b = gwave / (S_ * NH_);

  const float* base = qkv + (size_t)b * S_ * H3_;
  const int d0 = h * DK_ + lane;            // lane handles dims d0, d0+32
  const float q0 = base[(size_t)i * H3_ + d0];
  const float q1 = base[(size_t)i * H3_ + d0 + 32];

  float sc[18];
#pragma unroll
  for (int t = 0; t < 18; ++t) {
    int j = isBwd ? (i + t) : (i - 17 + t);
    float s = -1.0e9f;
    if (j >= 0 && j < S_) {                 // wave-uniform condition
      const float* kr = base + (size_t)j * H3_ + H_ + d0;
      float p = q0 * kr[0] + q1 * kr[32];
#pragma unroll
      for (int m = 16; m >= 1; m >>= 1) p += __shfl_xor(p, m, 32);
      s = p * 0.125f;                       // 1/sqrt(64)
    }
    sc[t] = s;
  }
  float mx = sc[0];
#pragma unroll
  for (int t = 1; t < 18; ++t) mx = fmaxf(mx, sc[t]);
  float den = 0.f, acc0 = 0.f, acc1 = 0.f;
#pragma unroll
  for (int t = 0; t < 18; ++t) {
    int j = isBwd ? (i + t) : (i - 17 + t);
    if (j >= 0 && j < S_) {
      float p = __expf(sc[t] - mx);
      den += p;
      const float* vr = base + (size_t)j * H3_ + 2 * H_ + d0;
      acc0 += p * vr[0];
      acc1 += p * vr[32];
    }
  }
  float inv = 1.0f / den;
  unsigned short* op = o + ((size_t)b * S_ + i) * H_ + d0;
  op[0]  = (unsigned short)f2bf(acc0 * inv);
  op[32] = (unsigned short)f2bf(acc1 * inv);
}

// =====================================================================
// combine: x = a[:, W+l, :] + sum_{kk=0..16} rw[kk] * a[:, off+kk+l, :]
// writes f32 (for gating / next layer) and bf16 (for highway GEMM)
// =====================================================================
__global__ __launch_bounds__(256) void combine_rel(
    const float* __restrict__ a, const float* __restrict__ rw,
    float* __restrict__ xo, unsigned short* __restrict__ xob, int offset)
{
  size_t idx = (size_t)blockIdx.x * 256 + threadIdx.x;
  int h = idx % H_;
  int l = (idx / H_) % L_;
  int b = idx / ((size_t)H_ * L_);
  const float* arow = a + ((size_t)b * S_ + offset + l) * H_ + h;
  float acc = a[((size_t)b * S_ + W_ + l) * H_ + h];
#pragma unroll
  for (int kk = 0; kk <= W_; ++kk) acc += rw[kk] * arow[(size_t)kk * H_];
  xo[idx]  = acc;
  xob[idx] = (unsigned short)f2bf(acc);
}

// =====================================================================
// highway gate: proj(B,L,2H) f32; x = sig(gate)*x + (1-sig(gate))*relu(nl)
// updates f32 x in place and refreshes the bf16 copy
// =====================================================================
__global__ __launch_bounds__(256) void highway_gate(
    float* __restrict__ x, unsigned short* __restrict__ xb16,
    const float* __restrict__ proj)
{
  size_t idx = (size_t)blockIdx.x * 256 + threadIdx.x;
  int h = idx % H_;
  size_t r = idx / H_;
  float nl = proj[r * H2_ + h];
  float gt = proj[r * H2_ + H_ + h];
  float g  = 1.0f / (1.0f + __expf(-gt));
  float v  = g * x[idx] + (1.0f - g) * fmaxf(nl, 0.0f);
  x[idx]   = v;
  xb16[idx] = (unsigned short)f2bf(v);
}

// =====================================================================
// out[layer] (B,L,2H) = concat(xf, xb) on last axis
// =====================================================================
__global__ __launch_bounds__(256) void write_out(
    const float* __restrict__ xf, const float* __restrict__ xb,
    float* __restrict__ out)
{
  size_t idx = (size_t)blockIdx.x * 256 + threadIdx.x;
  int h = idx % H_;
  size_t r = idx / H_;
  out[r * H2_ + h]      = xf[idx];
  out[r * H2_ + H_ + h] = xb[idx];
}

// =====================================================================
extern "C" void kernel_launch(void* const* d_in, const int* in_sizes, int n_in,
                              void* d_out, int out_size, void* d_ws, size_t ws_size,
                              hipStream_t stream)
{
  const float* inputs     = (const float*)d_in[0];
  // d_in[1] = masks (unused by reference)
  const float* fwd_attn_w = (const float*)d_in[2];
  const float* fwd_attn_b = (const float*)d_in[3];
  const float* bwd_attn_w = (const float*)d_in[4];
  const float* bwd_attn_b = (const float*)d_in[5];
  const float* fwd_pad    = (const float*)d_in[6];
  const float* bwd_pad    = (const float*)d_in[7];
  const float* fwd_rw     = (const float*)d_in[8];
  const float* bwd_rw     = (const float*)d_in[9];
  const float* fwd_hw_w   = (const float*)d_in[10];
  const float* fwd_hw_b   = (const float*)d_in[11];
  const float* bwd_hw_w   = (const float*)d_in[12];
  const float* bwd_hw_b   = (const float*)d_in[13];
  float* out = (float*)d_out;

  const size_t NBLH = (size_t)B_ * L_ * H_;          // 4,194,304
  const size_t NBSH = (size_t)B_ * S_ * H_;          // 4,456,448
  const size_t NQKV = (size_t)B_ * S_ * H3_;         // 13,369,344
  const size_t WATT = (size_t)NL_ * 4 * H_ * H_;     // 2,097,152
  const size_t WHWY = (size_t)NL_ * NHW_ * H2_ * H_; // 2,097,152

  // ---- workspace carve (256B aligned slabs) ----
  char* wsb = (char*)d_ws;
  auto carve = [&](size_t bytes) {
    char* p = wsb;
    wsb += (bytes + 255) & ~(size_t)255;
    return p;
  };
  float* xf    = (float*)carve(NBLH * 4);  // persistent fwd activations (f32)
  float* xb    = (float*)carve(NBLH * 4);  // persistent bwd activations (f32)
  float* qkv   = (float*)carve(NQKV * 4);  // QKV; reused as highway proj
  float* aproj = (float*)carve(NBSH * 4);  // O-projection output (f32)
  unsigned short* xpbf = (unsigned short*)carve(NBSH * 2);  // padded input (bf16)
  unsigned short* aobf = (unsigned short*)carve(NBSH * 2);  // attention out (bf16)
  unsigned short* xbbf = (unsigned short*)carve(NBLH * 2);  // highway input (bf16)
  unsigned short* wfa  = (unsigned short*)carve(WATT * 2);  // fwd attn weights bf16
  unsigned short* wba  = (unsigned short*)carve(WATT * 2);  // bwd attn weights bf16
  unsigned short* wfh  = (unsigned short*)carve(WHWY * 2);  // fwd hw weights bf16
  unsigned short* wbh  = (unsigned short*)carve(WHWY * 2);  // bwd hw weights bf16

  hipMemcpyAsync(xf, inputs, NBLH * sizeof(float), hipMemcpyDeviceToDevice, stream);
  hipMemcpyAsync(xb, inputs, NBLH * sizeof(float), hipMemcpyDeviceToDevice, stream);

  // ---- one-time (per call) weight conversion to bf16 ----
  cvt_bf16<<<(int)(WATT / 2048), 256, 0, stream>>>(fwd_attn_w, wfa);
  cvt_bf16<<<(int)(WATT / 2048), 256, 0, stream>>>(bwd_attn_w, wba);
  cvt_bf16<<<(int)(WHWY / 2048), 256, 0, stream>>>(fwd_hw_w, wfh);
  cvt_bf16<<<(int)(WHWY / 2048), 256, 0, stream>>>(bwd_hw_w, wbh);

  const int BS = B_ * S_;   // 8704 = 128*68
  const int BL = B_ * L_;   // 8192 = 128*64
  const int gPad  = (int)(NBSH / 256);           // 17408
  const int gElt  = (int)(NBLH / 256);           // 16384
  const int gAttn = (B_ * NH_ * S_ * 32) / 256;  // 8704

  for (int layer = 0; layer < NL_; ++layer) {
    for (int dir = 0; dir < 2; ++dir) {
      float* x = dir ? xb : xf;
      const bf16_t* attw = (const bf16_t*)((dir ? wba : wfa) + (size_t)layer * 4 * H_ * H_);
      const float* attb  = (dir ? bwd_attn_b : fwd_attn_b) + (size_t)layer * 4 * H_;
      const float* rw    = (dir ? bwd_rw     : fwd_rw)     + (size_t)layer * (W_ + 1);
      const bf16_t* hww  = (const bf16_t*)((dir ? wbh : wfh) + (size_t)layer * NHW_ * H2_ * H_);
      const float* hwb   = (dir ? bwd_hw_b : fwd_hw_b) + (size_t)layer * NHW_ * H2_;
      const float* pf    = fwd_pad + (size_t)layer * W_ * H_;  // front pad (both dirs)
      const float* pb    = bwd_pad + (size_t)layer * W_ * H_;  // back pad  (both dirs)

      // 1) pad (emit bf16 GEMM operand)
      pad_concat<<<gPad, 256, 0, stream>>>(x, pf, pb, xpbf);
      // 2) fused QKV projection: (8704 x 512) x (1536 x 512)^T
      gemm_bias_bf16<<<dim3(BS / 128, H3_ / 128), 256, 0, stream>>>(
          (const bf16_t*)xpbf, attw, attb, qkv, BS, H3_, H_);
      // 3) banded attention (bf16 out)
      band_attn<<<gAttn, 256, 0, stream>>>(qkv, aobf, dir);
      // 4) output projection
      gemm_bias_bf16<<<dim3(BS / 128, H_ / 128), 256, 0, stream>>>(
          (const bf16_t*)aobf, attw + (size_t)3 * H_ * H_, attb + 3 * H_,
          aproj, BS, H_, H_);
      // 5) center slice + 17-tap relative mix -> new x (f32 + bf16)
      combine_rel<<<gElt, 256, 0, stream>>>(aproj, rw, x, xbbf, dir ? W_ : 0);
      // 6) highway x2
      for (int j = 0; j < NHW_; ++j) {
        gemm_bias_bf16<<<dim3(BL / 128, H2_ / 128), 256, 0, stream>>>(
            (const bf16_t*)xbbf, hww + (size_t)j * H2_ * H_,
            hwb + (size_t)j * H2_, qkv, BL, H2_, H_);
        highway_gate<<<gElt, 256, 0, stream>>>(x, xbbf, qkv);
      }
    }
    // 7) layer output
    write_out<<<gElt, 256, 0, stream>>>(xf, xb, out + (size_t)layer * B_ * L_ * H2_);
  }
}
